// GPT_40948218200709
// MI455X (gfx1250) — matrix operand order repaired and checked by reading, not compiled, not run on hardware
//
#include <hip/hip_runtime.h>

// ---------------- problem constants (GPT-2 124M, B=1) ----------------
#define Tt   1024
#define Cc   768
#define Hh   12
#define Dd   64
#define Ll   12
#define Vv   50257

#define NB   64   // GEMM N tile per block
#define KB   32   // GEMM K step (WMMA K)
#define MB   256  // GEMM M tile per block (8 waves x 2 x 16 rows)

#ifndef USE_TDM
#define USE_TDM 1  // Tensor Data Mover copy of the attention V tile
#endif
#define VPITCH 128 // bytes per V row in LDS (reads are row-wise: conflict-free)

typedef __attribute__((ext_vector_type(16))) __bf16 v16bf;
typedef __attribute__((ext_vector_type(8)))  float  v8f;
typedef __attribute__((ext_vector_type(4)))  unsigned v4u;
typedef __attribute__((ext_vector_type(8)))  unsigned v8u;

static __device__ __forceinline__ __bf16 f2bf(float f) { return (__bf16)f; }

static __device__ __forceinline__ v8f wmma_bf16(v16bf a, v16bf b, v8f c) {
    return __builtin_amdgcn_wmma_f32_16x16x32_bf16(false, a, false, b, (short)0, c,
                                                   false, false);
}

static __device__ __forceinline__ float gelu_tanh(float x) {
    const float k = 0.7978845608028654f;           // sqrt(2/pi)
    return 0.5f * x * (1.0f + tanhf(k * (x + 0.044715f * x * x * x)));
}

// A-fragment K offset for element i (16-bit A 16x32 layout):
// lanes 0-15 hold K 0..7 and 16..23; lanes 16-31 hold K 8..15 and 24..31.
static __device__ __forceinline__ int a_koff(int i, int akb) {
    return akb + (i < 8 ? i : i + 8);
}

// A fragment from bf16: elements 0-7 at base[0..7], 8-15 at base[16..23].
static __device__ __forceinline__ v16bf load_afrag(const __bf16* base) {
    union { v16bf v; uint4 q[2]; } u;
    u.q[0] = *(const uint4*)(base);
    u.q[1] = *(const uint4*)(base + 16);
    return u.v;
}

// 16 consecutive bf16 (for row-contiguous K^T B-fragments).
static __device__ __forceinline__ v16bf load_contig16(const __bf16* base) {
    union { v16bf v; uint4 q[2]; } u;
    u.q[0] = *(const uint4*)(base);
    u.q[1] = *(const uint4*)(base + 8);
    return u.v;
}

#if USE_TDM
// TDM: DMA a rows x cols 2D bf16 tile (row pitch src_pitch elements) from
// global memory into LDS at lds_off. D# packing per CDNA5 ISA:
// group0: count=1 | lds_addr | global_addr(57b) | type=2
// group1: data_size=1(2B), tensor_dim0=src_pitch, tensor_dim1=rows,
//         tile_dim0=cols, tile_dim1=rows, dim0_stride=src_pitch.
static __device__ __forceinline__ void tdm_load_2d(unsigned lds_off,
                                                   const __bf16* gaddr,
                                                   unsigned rows, unsigned cols,
                                                   unsigned src_pitch) {
    unsigned long long ga = (unsigned long long)(uintptr_t)gaddr;
    v4u g0;
    g0[0] = 1u;                                            // count=1 (valid D#)
    g0[1] = lds_off;                                       // lds_addr (bytes)
    g0[2] = (unsigned)ga;                                  // global_addr[31:0]
    g0[3] = ((unsigned)(ga >> 32) & 0x01FFFFFFu) | (2u << 30); // [56:32]|type=2
    v8u g1;
    g1[0] = (1u << 16);                                    // data_size=2B
    g1[1] = (src_pitch & 0xFFFFu) << 16;                   // tensor_dim0 lo
    g1[2] = ((src_pitch >> 16) & 0xFFFFu) | ((rows & 0xFFFFu) << 16); // hi|dim1
    g1[3] = (cols & 0xFFFFu) << 16;                        // tile_dim0
    g1[4] = rows & 0xFFFFu;                                // tile_dim1 (dim2=0)
    g1[5] = src_pitch;                                     // dim0_stride lo32
    g1[6] = 0u;
    g1[7] = 0u;
    asm volatile("tensor_load_to_lds %0, %1" :: "s"(g0), "s"(g1) : "memory");
}
#endif

// ---------------------------------------------------------------------
// Embedding: x[t,c] = wte[idx[t],c] + wpe[t,c]   (fp32 residual stream)
// ---------------------------------------------------------------------
__global__ __launch_bounds__(256) void embed_k(const int* __restrict__ idx,
                                               const float* __restrict__ wte,
                                               const float* __restrict__ wpe,
                                               float* __restrict__ x) {
    int i = blockIdx.x * blockDim.x + threadIdx.x;
    if (i >= Tt * Cc) return;
    int t = i / Cc, c = i - t * Cc;
    x[i] = wte[(size_t)idx[t] * Cc + c] + wpe[i];
}

// ---------------------------------------------------------------------
// LayerNorm over C=768, bf16 output. One block (256 threads) per row.
// ---------------------------------------------------------------------
__global__ __launch_bounds__(256) void layernorm_k(const float* __restrict__ x,
                                                   const float* __restrict__ w,
                                                   const float* __restrict__ b,
                                                   __bf16* __restrict__ out) {
    __shared__ float ssum[256];
    __shared__ float ssq[256];
    const int row = blockIdx.x;
    const float* xr = x + (size_t)row * Cc;
    float s = 0.f, q = 0.f;
    for (int c = threadIdx.x; c < Cc; c += 256) {
        float v = xr[c];
        s += v;
        q += v * v;
    }
    ssum[threadIdx.x] = s;
    ssq[threadIdx.x]  = q;
    __syncthreads();
    for (int off = 128; off > 0; off >>= 1) {
        if (threadIdx.x < off) {
            ssum[threadIdx.x] += ssum[threadIdx.x + off];
            ssq[threadIdx.x]  += ssq[threadIdx.x + off];
        }
        __syncthreads();
    }
    const float mu   = ssum[0] * (1.0f / Cc);
    const float var  = ssq[0] * (1.0f / Cc) - mu * mu;
    const float rstd = rsqrtf(var + 1e-5f);
    for (int c = threadIdx.x; c < Cc; c += 256)
        out[(size_t)row * Cc + c] = f2bf((xr[c] - mu) * rstd * w[c] + b[c]);
}

// ---------------------------------------------------------------------
// GEMM: out = act(A_bf16[M,K] @ B_f32[K,N] + bias [+ residual])
// Block = 256 threads (8 waves), tile 256(M) x 64(N), K stepped by 32.
// B tile staged f32 -> bf16 -> LDS in B-fragment order, double buffered.
// 8 WMMAs per wave per K step. grid = ((N+63)/64, M/256)
// ---------------------------------------------------------------------
__global__ __launch_bounds__(256) void gemm_bf16_wmma(
        const __bf16* __restrict__ A, int lda,
        const float*  __restrict__ B, int ldb,
        const float*  __restrict__ bias,
        const float*  __restrict__ residual,     // stride ldc, may alias outF
        float* __restrict__ outF,                // fp32 output (or null)
        __bf16* __restrict__ outH,               // bf16 output (or null)
        int ldc, int M, int N, int K, int fuse_gelu) {
    __shared__ v16bf Fs[2][4][32];               // [buf][n-frag][lane]

    const int tid  = threadIdx.x;
    const int lane = tid & 31;
    const int wave = tid >> 5;
    const int n0   = blockIdx.x * NB;
    const int m0a  = blockIdx.y * MB + wave * 16;
    const int m0b  = m0a + 128;

    const int am     = lane & 15;
    const int akb    = (lane >> 4) * 8;          // 0 | 8
    const int rowoff = (lane >> 4) << 3;         // 0 | 8

    const int skrow = tid >> 3;                  // 0..31
    const int sncol = (tid & 7) * 8;             // 0..56
    const bool colsafe = (n0 + NB <= N);

    v8f acc[2][4];
#pragma unroll
    for (int t = 0; t < 2; ++t)
#pragma unroll
        for (int j = 0; j < 4; ++j)
            acc[t][j] = (v8f){0.f, 0.f, 0.f, 0.f, 0.f, 0.f, 0.f, 0.f};

    auto stage = [&](int k0, int nb) {
        const float* src = B + (size_t)(k0 + skrow) * ldb + n0 + sncol;
        if (k0 + 2 * KB < K)
            __builtin_prefetch(src + (size_t)KB * ldb, 0, 1);
        float v[8];
        if (colsafe) {
#pragma unroll
            for (int ii = 0; ii < 8; ++ii) v[ii] = src[ii];
        } else {
            const float* rowp = B + (size_t)(k0 + skrow) * ldb;
#pragma unroll
            for (int ii = 0; ii < 8; ++ii) {
                int col = n0 + sncol + ii;
                v[ii] = rowp[col < N ? col : N - 1];   // clamp: no exec dance
            }
        }
        const int lnhalf = (skrow & 8) << 1;       // 0 | 16 (K half)
        const int ielem  = (skrow & 7) | ((skrow & 16) >> 1);
#pragma unroll
        for (int ii = 0; ii < 8; ++ii) {
            const int n = sncol + ii;
            ((__bf16*)&Fs[nb][n >> 4][(n & 15) | lnhalf])[ielem] = f2bf(v[ii]);
        }
    };

    int buf = 0;
    stage(0, 0);
    __syncthreads();

    const __bf16* arow0 = A + (size_t)(m0a + am) * lda + akb;
    const __bf16* arow1 = A + (size_t)(m0b + am) * lda + akb;

    for (int k0 = 0; k0 < K; k0 += KB) {
        if (k0 + KB < K) stage(k0 + KB, buf ^ 1);
        const v16bf a0 = load_afrag(arow0 + k0);
        const v16bf a1 = load_afrag(arow1 + k0);
#pragma unroll
        for (int j = 0; j < 4; ++j) {
            const v16bf bf = Fs[buf][j][lane];     // 2x ds_load_b128
            acc[0][j] = wmma_bf16(a0, bf, acc[0][j]);
            acc[1][j] = wmma_bf16(a1, bf, acc[1][j]);
        }
        __syncthreads();
        buf ^= 1;
    }

#pragma unroll
    for (int t = 0; t < 2; ++t) {
        const int mrow = (t ? m0b : m0a) + rowoff;
#pragma unroll
        for (int j = 0; j < 4; ++j) {
            const int col = n0 + j * 16 + (lane & 15);
            if (col < N) {
#pragma unroll
                for (int i = 0; i < 8; ++i) {
                    float v = acc[t][j][i];
                    if (bias)      v += bias[col];
                    const size_t o = (size_t)(mrow + i) * ldc + col;
                    if (residual)  v += residual[o];
                    if (fuse_gelu) v = gelu_tanh(v);
                    if (outF) outF[o] = v;
                    if (outH) outH[o] = f2bf(v);
                }
            }
        }
    }
}

// ---------------------------------------------------------------------
// Causal flash attention, one wave per (16-row query tile, head).
// qkv is bf16 [T, 3C]. Key tiles in PAIRS (32 keys/step):
//   S tiles : Q A-frags + row-contiguous K^T B-frags (16B vector loads),
//             4 WMMAs; 1/8 scale applied post-WMMA.
//   softmax : online, shfl_xor row reductions in 16-lane halves.
//   V tile  : 32x64 bf16 DMA'd to LDS by the Tensor Data Mover while S &
//             softmax run; synced with s_wait_tensorcnt; conflict-free
//             row-wise fragment reads. O += P V via 4 WMMAs.
// grid = (T/16, H), block = 32.
// ---------------------------------------------------------------------
__global__ __launch_bounds__(32) void attn_wmma(const __bf16* __restrict__ qkv,
                                                __bf16* __restrict__ y16) {
    const int lane = threadIdx.x & 31;
    const int qt   = blockIdx.x;
    const int h    = blockIdx.y;
    const int m0   = qt * 16;
    const int qcol = h * Dd;
    const int kcol = Cc + h * Dd;
    const int vcol = 2 * Cc + h * Dd;

    const int am     = lane & 15;
    const int akb    = (lane >> 4) * 8;
    const int bn     = lane & 15;
    const int bkb    = (lane >> 4) * 16;
    const int rowoff = (lane >> 4) << 3;

    __shared__ float Pt[16][33];                 // 16 x 32 P tile (+1 pad)
#if USE_TDM
    __shared__ __align__(16) char Vsm[32 * VPITCH];  // V tile, bf16 rows
    const unsigned vsm_off = (unsigned)(uintptr_t)(void*)Vsm; // addr[31:0]=LDS off
#endif

    // Q fragments (raw bits; softmax scale applied to S instead)
    const __bf16* qrow = qkv + (size_t)(m0 + am) * (3 * Cc) + qcol + akb;
    const v16bf aq0 = load_afrag(qrow);
    const v16bf aq1 = load_afrag(qrow + 32);

    float run_m[8], run_l[8];
#pragma unroll
    for (int i = 0; i < 8; ++i) { run_m[i] = -1e30f; run_l[i] = 0.f; }
    v8f accO[4];
#pragma unroll
    for (int j = 0; j < 4; ++j) accO[j] = (v8f){0.f, 0.f, 0.f, 0.f, 0.f, 0.f, 0.f, 0.f};

    const int npair = (qt >> 1) + 1;
    for (int jp = 0; jp < npair; ++jp) {
        const int t0 = jp * 2;                   // key tiles t0, t0+1 (rows < T)

#if USE_TDM
        // async: V tile (rows t0*16 .. +31, 64 cols) -> LDS
        tdm_load_2d(vsm_off, qkv + (size_t)(t0 * 16) * (3 * Cc) + vcol,
                    32u, 64u, (unsigned)(3 * Cc));
#endif

        // ---- S tiles for both key tiles (K^T frags are row-contiguous) ----
        v8f s[2];
#pragma unroll
        for (int tt = 0; tt < 2; ++tt) {
            const __bf16* kbase =
                qkv + (size_t)((t0 + tt) * 16 + bn) * (3 * Cc) + kcol + akb;
            const v16bf bk0 = load_afrag(kbase);
            const v16bf bk1 = load_afrag(kbase + 32);
            v8f sv = (v8f){0.f, 0.f, 0.f, 0.f, 0.f, 0.f, 0.f, 0.f};
            sv = wmma_bf16(aq0, bk0, sv);
            sv = wmma_bf16(aq1, bk1, sv);
            s[tt] = sv;
        }

        // ---- causal mask + online softmax over 32 columns ----
        float p0[8], p1[8], alpha[8];
#pragma unroll
        for (int i = 0; i < 8; ++i) {
            const int grow = m0 + rowoff + i;
            const int c0   = t0 * 16 + (lane & 15);
            const float sv0 = (c0      <= grow) ? s[0][i] * 0.125f : -1e30f;
            const float sv1 = (c0 + 16 <= grow) ? s[1][i] * 0.125f : -1e30f;
            float mx = fmaxf(sv0, sv1);
            mx = fmaxf(mx, __shfl_xor(mx, 1, 32));
            mx = fmaxf(mx, __shfl_xor(mx, 2, 32));
            mx = fmaxf(mx, __shfl_xor(mx, 4, 32));
            mx = fmaxf(mx, __shfl_xor(mx, 8, 32));
            const float nm = fmaxf(run_m[i], mx);
            alpha[i] = __expf(run_m[i] - nm);
            run_m[i] = nm;
            p0[i] = __expf(sv0 - nm);
            p1[i] = __expf(sv1 - nm);
            float sum = p0[i] + p1[i];
            sum += __shfl_xor(sum, 1, 32);
            sum += __shfl_xor(sum, 2, 32);
            sum += __shfl_xor(sum, 4, 32);
            sum += __shfl_xor(sum, 8, 32);
            run_l[i] = alpha[i] * run_l[i] + sum;
        }
#pragma unroll
        for (int c4 = 0; c4 < 4; ++c4)
#pragma unroll
            for (int i = 0; i < 8; ++i) accO[c4][i] *= alpha[i];

        // ---- P: D-layout -> LDS -> A-layout (full K=32) ----
        __syncthreads();
#pragma unroll
        for (int i = 0; i < 8; ++i) {
            Pt[rowoff + i][lane & 15]        = p0[i];
            Pt[rowoff + i][16 + (lane & 15)] = p1[i];
        }
        __syncthreads();

        v16bf aP;
#pragma unroll
        for (int i = 0; i < 16; ++i) aP[i] = f2bf(Pt[am][a_koff(i, akb)]);

#if USE_TDM
        __builtin_amdgcn_s_wait_tensorcnt(0);    // V tile resident in LDS
#endif

        // ---- O += P V ----
#pragma unroll
        for (int c4 = 0; c4 < 4; ++c4) {
            v16bf bv;
#pragma unroll
            for (int i = 0; i < 16; ++i) {
                const int kd = a_koff(i, akb);   // 0..31
#if USE_TDM
                bv[i] = *(const __bf16*)&Vsm[(size_t)kd * VPITCH +
                                             2 * (c4 * 16 + bn)];
#else
                bv[i] = qkv[(size_t)(t0 * 16 + kd) * (3 * Cc) + vcol +
                            c4 * 16 + bn];
#endif
            }
            accO[c4] = wmma_bf16(aP, bv, accO[c4]);
        }
    }

    // ---- epilogue: y16[t, h*64+d] = bf16(O / l) ----
#pragma unroll
    for (int c4 = 0; c4 < 4; ++c4)
#pragma unroll
        for (int i = 0; i < 8; ++i) {
            const int row = m0 + rowoff + i;
            const int col = h * Dd + c4 * 16 + (lane & 15);
            y16[(size_t)row * Cc + col] = f2bf(accO[c4][i] / run_l[i]);
        }
}

// ---------------------------------------------------------------------
// Host-side orchestration
// ---------------------------------------------------------------------
extern "C" void kernel_launch(void* const* d_in, const int* in_sizes, int n_in,
                              void* d_out, int out_size, void* d_ws, size_t ws_size,
                              hipStream_t stream) {
    (void)in_sizes; (void)n_in; (void)out_size; (void)ws_size;

    const int*   idx    = (const int*)  d_in[0];
    const float* wte    = (const float*)d_in[1];
    const float* wpe    = (const float*)d_in[2];
    const float* ln1_w  = (const float*)d_in[3];
    const float* ln1_b  = (const float*)d_in[4];
    const float* w_attn = (const float*)d_in[5];
    const float* b_attn = (const float*)d_in[6];
    const float* w_proj = (const float*)d_in[7];
    const float* b_proj = (const float*)d_in[8];
    const float* ln2_w  = (const float*)d_in[9];
    const float* ln2_b  = (const float*)d_in[10];
    const float* w_fc   = (const float*)d_in[11];
    const float* b_fc   = (const float*)d_in[12];
    const float* w_fc2  = (const float*)d_in[13];
    const float* b_fc2  = (const float*)d_in[14];
    const float* lnf_w  = (const float*)d_in[15];
    const float* lnf_b  = (const float*)d_in[16];
    const float* lm_head= (const float*)d_in[17];
    float* logits = (float*)d_out;

    char* w = (char*)d_ws;                                   // 16B-aligned offsets
    float*  xb     = (float*)w;   w += (size_t)Tt * Cc * 4;      // residual, f32
    __bf16* hb16   = (__bf16*)w;  w += (size_t)Tt * Cc * 2;      // LN out, bf16
    __bf16* qkvh   = (__bf16*)w;  w += (size_t)Tt * 3 * Cc * 2;  // qkv, bf16
    __bf16* yb16   = (__bf16*)w;  w += (size_t)Tt * Cc * 2;      // attn out, bf16
    __bf16* mlpb16 = (__bf16*)w;                                 // gelu out, bf16

    const dim3 blk32(32), blk256(256);
    const dim3 gq((3 * Cc + NB - 1) / NB, Tt / MB);
    const dim3 gp((Cc + NB - 1) / NB, Tt / MB);
    const dim3 gf((4 * Cc + NB - 1) / NB, Tt / MB);
    const dim3 gl((Vv + NB - 1) / NB, Tt / MB);

    embed_k<<<(Tt * Cc + 255) / 256, blk256, 0, stream>>>(idx, wte, wpe, xb);

    for (int l = 0; l < Ll; ++l) {
        layernorm_k<<<Tt, blk256, 0, stream>>>(xb, ln1_w + (size_t)l * Cc,
                                               ln1_b + (size_t)l * Cc, hb16);
        // qkv(bf16) = h @ w_attn + b_attn
        gemm_bf16_wmma<<<gq, blk256, 0, stream>>>(
            hb16, Cc, w_attn + (size_t)l * Cc * 3 * Cc, 3 * Cc,
            b_attn + (size_t)l * 3 * Cc, nullptr,
            nullptr, qkvh, 3 * Cc, Tt, 3 * Cc, Cc, 0);

        attn_wmma<<<dim3(Tt / 16, Hh), blk32, 0, stream>>>(qkvh, yb16);

        // x(f32) = x + y @ w_proj + b_proj
        gemm_bf16_wmma<<<gp, blk256, 0, stream>>>(
            yb16, Cc, w_proj + (size_t)l * Cc * Cc, Cc,
            b_proj + (size_t)l * Cc, xb,
            xb, nullptr, Cc, Tt, Cc, Cc, 0);

        layernorm_k<<<Tt, blk256, 0, stream>>>(xb, ln2_w + (size_t)l * Cc,
                                               ln2_b + (size_t)l * Cc, hb16);
        // mlp(bf16) = gelu(h @ w_fc + b_fc)
        gemm_bf16_wmma<<<gf, blk256, 0, stream>>>(
            hb16, Cc, w_fc + (size_t)l * Cc * 4 * Cc, 4 * Cc,
            b_fc + (size_t)l * 4 * Cc, nullptr,
            nullptr, mlpb16, 4 * Cc, Tt, 4 * Cc, Cc, 1);
        // x(f32) = x + mlp @ w_fc2 + b_fc2
        gemm_bf16_wmma<<<gp, blk256, 0, stream>>>(
            mlpb16, 4 * Cc, w_fc2 + (size_t)l * 4 * Cc * Cc, Cc,
            b_fc2 + (size_t)l * Cc, xb,
            xb, nullptr, Cc, Tt, Cc, 4 * Cc, 0);
    }

    layernorm_k<<<Tt, blk256, 0, stream>>>(xb, lnf_w, lnf_b, hb16);
    // logits(f32) = h @ lm_head   (N = 50257, ragged tile clamped+guarded)
    gemm_bf16_wmma<<<gl, blk256, 0, stream>>>(
        hb16, Cc, lm_head, Vv, nullptr, nullptr,
        logits, nullptr, Vv, Tt, Vv, Cc, 0);
}
